// MoEMLP_58179626991687
// MI455X (gfx1250) — matrix operand order repaired and checked by reading, not compiled, not run on hardware
//
#include <hip/hip_runtime.h>
#include <hip/hip_bf16.h>

// MoE MLP (T=2048, D=1024, H=2048, E=8, top-2) for gfx1250 (MI455X).
// bf16 WMMA 16x16x32 + f32 accumulation. Weights/x pre-converted to bf16 in
// workspace (one ~9us memory pass at 23.3 TB/s); router builds per-expert
// compacted token lists so the expert kernel only computes routed (token,
// expert) pairs -> ~4x less WMMA work than dense, mathematically identical to
// the reference (unrouted experts have zero combine weight).

#define TOK   2048
#define DIM   1024
#define HID   2048
#define NEXP  8
#define TM    32          // tokens (slots) per workgroup tile
#define HC    64          // hidden-chunk per iteration
#define XP    1032        // LDS pitch for x tile (bf16 elems): 16B-aligned, bank-staggered
#define HP    72          // LDS pitch for h chunk (bf16 elems): 16B-aligned

typedef __attribute__((ext_vector_type(16))) __bf16 v16bf;
typedef __attribute__((ext_vector_type(8)))  float  v8f;

union FragU { uint4 u[2]; v16bf v; };

// Load 16 consecutive f32 and convert to a bf16 fragment (fallback path).
__device__ __forceinline__ v16bf load_b16_from_f32(const float* __restrict__ p) {
    const float4* q = (const float4*)p;
    float4 f0 = q[0], f1 = q[1], f2 = q[2], f3 = q[3];
    v16bf b;
    b[0]  = (__bf16)f0.x; b[1]  = (__bf16)f0.y; b[2]  = (__bf16)f0.z; b[3]  = (__bf16)f0.w;
    b[4]  = (__bf16)f1.x; b[5]  = (__bf16)f1.y; b[6]  = (__bf16)f1.z; b[7]  = (__bf16)f1.w;
    b[8]  = (__bf16)f2.x; b[9]  = (__bf16)f2.y; b[10] = (__bf16)f2.z; b[11] = (__bf16)f2.w;
    b[12] = (__bf16)f3.x; b[13] = (__bf16)f3.y; b[14] = (__bf16)f3.z; b[15] = (__bf16)f3.w;
    return b;
}

// Load 16 consecutive bf16 (32 bytes) as a fragment (fast path).
__device__ __forceinline__ v16bf load_b16_direct(const __bf16* __restrict__ p) {
    FragU f;
    f.u[0] = *(const uint4*)p;
    f.u[1] = *(const uint4*)(p + 8);
    return f.v;
}

__global__ void moe_zero_kernel(float* __restrict__ p, int n) {
    int i = blockIdx.x * 256 + threadIdx.x;
    if (i < n) p[i] = 0.0f;
}

// f32 -> bf16 pre-conversion, 4 elements/thread (float4 in, 8B out).
__global__ void moe_cvt_kernel(const float* __restrict__ in,
                               __bf16* __restrict__ out, int n4) {
    int i = blockIdx.x * 256 + threadIdx.x;
    if (i < n4) {
        float4 f = ((const float4*)in)[i];
        union { __bf16 h[4]; uint2 u; } p;
        p.h[0] = (__bf16)f.x; p.h[1] = (__bf16)f.y;
        p.h[2] = (__bf16)f.z; p.h[3] = (__bf16)f.w;
        ((uint2*)out)[i] = p.u;
    }
}

// One wave32 per token: router logits -> softmax -> top-2 -> renormalized
// weights. Writes dense wgt[token][expert] (fallback path) and, if
// buildLists, per-expert compacted (token, weight) lists for the fast path.
__global__ void __launch_bounds__(256) moe_router_kernel(
        const float* __restrict__ x, const float* __restrict__ Wr,
        float* __restrict__ wgt,
        int* __restrict__ cnt, int* __restrict__ idxl, float* __restrict__ wl,
        int buildLists) {
    const int t    = blockIdx.x * 8 + (threadIdx.x >> 5);
    const int lane = threadIdx.x & 31;
    const float* xr = x + (size_t)t * DIM;

    float acc[NEXP];
#pragma unroll
    for (int e = 0; e < NEXP; ++e) acc[e] = 0.0f;

    for (int d = lane; d < DIM; d += 32) {
        float xv = xr[d];
#pragma unroll
        for (int e = 0; e < NEXP; ++e) acc[e] += xv * Wr[e * DIM + d];
    }
#pragma unroll
    for (int e = 0; e < NEXP; ++e) {
        acc[e] += __shfl_xor(acc[e], 16);
        acc[e] += __shfl_xor(acc[e], 8);
        acc[e] += __shfl_xor(acc[e], 4);
        acc[e] += __shfl_xor(acc[e], 2);
        acc[e] += __shfl_xor(acc[e], 1);
    }
    float m = acc[0];
#pragma unroll
    for (int e = 1; e < NEXP; ++e) m = fmaxf(m, acc[e]);
    float p[NEXP], s = 0.0f;
#pragma unroll
    for (int e = 0; e < NEXP; ++e) { p[e] = __expf(acc[e] - m); s += p[e]; }
    float inv = 1.0f / s;
#pragma unroll
    for (int e = 0; e < NEXP; ++e) p[e] *= inv;
    // top-2, lowest index wins ties (matches lax.top_k)
    int i1 = 0;
#pragma unroll
    for (int e = 1; e < NEXP; ++e) if (p[e] > p[i1]) i1 = e;
    int i2 = (i1 == 0) ? 1 : 0;
#pragma unroll
    for (int e = 0; e < NEXP; ++e) if (e != i1 && p[e] > p[i2]) i2 = e;
    float rn = 1.0f / (p[i1] + p[i2] + 1e-8f);
    float w1 = p[i1] * rn, w2 = p[i2] * rn;
    if (lane < NEXP) {
        int e = lane;
        wgt[(size_t)t * NEXP + e] =
            (e == i1) ? w1 : ((e == i2) ? w2 : 0.0f);
    }
    if (buildLists && lane == 0) {
        int s1 = atomicAdd(&cnt[i1], 1);
        idxl[i1 * TOK + s1] = t;  wl[i1 * TOK + s1] = w1;
        int s2 = atomicAdd(&cnt[i2], 1);
        idxl[i2 * TOK + s2] = t;  wl[i2 * TOK + s2] = w2;
    }
}

// Fused expert MLP. PRE=true: gathered token lists + bf16 weights/x in ws;
// blocks past cnt[e] exit; partial tiles padded with (token 0, weight 0).
// PRE=false: dense tokens, f32 weights converted on the fly.
// Per tile: loop H in chunks of 64; phase1 h = relu(x@Wfc^T)^2 (bf16, LDS),
// phase2 acc += h @ Wproj^T; epilogue scales by combine weight + atomicAdd.
template <bool PRE>
__global__ void __launch_bounds__(256) moe_expert_kernel(
        const float*  __restrict__ x,
        const __bf16* __restrict__ x16,
        const float*  __restrict__ Wfc,    // [E][H][D] f32
        const float*  __restrict__ Wp,     // [E][D][H] f32
        const __bf16* __restrict__ Wfc16,  // [E][H][D] bf16
        const __bf16* __restrict__ Wp16,   // [E][D][H] bf16
        const float*  __restrict__ wgt,    // [T][E]   (dense path)
        const int*    __restrict__ cnt,    // [E]      (gather path)
        const int*    __restrict__ idxl,   // [E][T]
        const float*  __restrict__ wl,     // [E][T]
        float* __restrict__ out) {         // [T][D]
    const int e  = blockIdx.y;
    const int s0 = blockIdx.x * TM;        // first slot of this tile
    const int n_e = PRE ? cnt[e] : TOK;
    if (s0 >= n_e) return;                 // uniform early-exit (empty tile)

    __shared__ __bf16 xbf[TM * XP];
    __shared__ __bf16 hbuf[TM * HP];

    const int tid  = threadIdx.x;
    const int w    = tid >> 5;
    const int lane = tid & 31;
    const int hi   = lane >> 4;   // K-half selector for A/B fragments
    const int ln   = lane & 15;   // M (A) / N (B,C) index
    const int eoff = e * TOK;

    // ---- stage x tile [32 x 1024] as bf16 into LDS (gathered if PRE) ----
    if constexpr (PRE) {
        for (int idx = tid; idx < TM * (DIM / 8); idx += 256) {
            int r   = idx >> 7;            // DIM/8 == 128 uint4 per row
            int c8  = idx & 127;
            int sl  = s0 + r;
            int tok = (sl < n_e) ? idxl[eoff + sl] : 0;
            uint4 v = ((const uint4*)(x16 + (size_t)tok * DIM))[c8];
            *(uint4*)&xbf[r * XP + c8 * 8] = v;
        }
    } else {
        const float* xt = x + (size_t)s0 * DIM;
        for (int idx = tid; idx < TM * (DIM / 4); idx += 256) {
            int r  = idx >> 8;             // DIM/4 == 256 float4 per row
            int c4 = idx & 255;
            float4 f = ((const float4*)(xt + (size_t)r * DIM))[c4];
            __bf16* d = &xbf[r * XP + c4 * 4];
            d[0] = (__bf16)f.x; d[1] = (__bf16)f.y;
            d[2] = (__bf16)f.z; d[3] = (__bf16)f.w;
        }
    }

    const int m1 = w & 1;            // phase1: token sub-tile
    const int jt = w >> 1;           // phase1: hidden sub-tile (0..3) in chunk
    const int m2 = w & 1;            // phase2: token sub-tile
    const int dg = (w >> 1) * 16;    // phase2: first of 16 owned d-tiles

    v8f acc[16];
#pragma unroll
    for (int i = 0; i < 16; ++i)
        acc[i] = (v8f){0.f, 0.f, 0.f, 0.f, 0.f, 0.f, 0.f, 0.f};

    const float*  wfc_f = Wfc   + (size_t)e * HID * DIM;
    const __bf16* wfc_h = Wfc16 + (size_t)e * HID * DIM;
    const float*  wp_f  = Wp    + (size_t)e * DIM * HID;
    const __bf16* wp_h  = Wp16  + (size_t)e * DIM * HID;
    const int arow1 = (m1 * 16 + ln) * XP;
    const int arow2 = (m2 * 16 + ln) * HP;

    for (int hc = 0; hc < HID / HC; ++hc) {
        __syncthreads();   // xbf ready (iter 0); prev phase2 hbuf reads done
        // ---- phase 1: one 16x16 h-tile per wave, K = D = 1024 ----
        {
            const int j0 = hc * HC + jt * 16;
            v8f c = (v8f){0.f, 0.f, 0.f, 0.f, 0.f, 0.f, 0.f, 0.f};
#pragma unroll 4
            for (int ks = 0; ks < DIM / 32; ++ks) {
                const int k0 = ks * 32;
                FragU a;
                a.u[0] = *(const uint4*)&xbf[arow1 + k0 + hi * 16];
                a.u[1] = *(const uint4*)&xbf[arow1 + k0 + hi * 16 + 8];
                v16bf b;
                if constexpr (PRE)
                    b = load_b16_direct(wfc_h + (size_t)(j0 + ln) * DIM + k0 + hi * 16);
                else
                    b = load_b16_from_f32(wfc_f + (size_t)(j0 + ln) * DIM + k0 + hi * 16);
                c = __builtin_amdgcn_wmma_f32_16x16x32_bf16(
                        false, a.v, false, b, (short)0, c, false, false);
            }
            // relu^2 -> bf16 -> LDS   (C layout: M = 8*hi + g, N = ln)
#pragma unroll
            for (int g = 0; g < 8; ++g) {
                float r = fmaxf(c[g], 0.0f);
                hbuf[(m1 * 16 + hi * 8 + g) * HP + jt * 16 + ln] = (__bf16)(r * r);
            }
        }
        __syncthreads();
        // ---- phase 2: 16 output d-tiles per wave, K = chunk of 64 ----
#pragma unroll
        for (int ks = 0; ks < 2; ++ks) {
            const int kk = ks * 32;
            FragU a;
            a.u[0] = *(const uint4*)&hbuf[arow2 + kk + hi * 16];
            a.u[1] = *(const uint4*)&hbuf[arow2 + kk + hi * 16 + 8];
            const int hcol = hc * HC + kk + hi * 16;
#pragma unroll
            for (int dt = 0; dt < 16; ++dt) {
                const int d0 = (dg + dt) * 16;
                v16bf b;
                if constexpr (PRE)
                    b = load_b16_direct(wp_h + (size_t)(d0 + ln) * HID + hcol);
                else
                    b = load_b16_from_f32(wp_f + (size_t)(d0 + ln) * HID + hcol);
                acc[dt] = __builtin_amdgcn_wmma_f32_16x16x32_bf16(
                        false, a.v, false, b, (short)0, acc[dt], false, false);
            }
        }
    }

    // ---- combine weight + scatter-accumulate into output ----
    int   tokg[8];
    float wtr[8];
#pragma unroll
    for (int g = 0; g < 8; ++g) {
        int sl = s0 + m2 * 16 + hi * 8 + g;
        if constexpr (PRE) {
            bool v  = sl < n_e;
            tokg[g] = v ? idxl[eoff + sl] : 0;
            wtr[g]  = v ? wl[eoff + sl] : 0.0f;   // pad -> adds 0.0
        } else {
            tokg[g] = sl;
            wtr[g]  = wgt[(size_t)sl * NEXP + e];
        }
    }
#pragma unroll
    for (int dt = 0; dt < 16; ++dt) {
        const int d0 = (dg + dt) * 16 + ln;
#pragma unroll
        for (int g = 0; g < 8; ++g) {
            atomicAdd(out + (size_t)tokg[g] * DIM + d0, acc[dt][g] * wtr[g]);
        }
    }
}

extern "C" void kernel_launch(void* const* d_in, const int* in_sizes, int n_in,
                              void* d_out, int out_size, void* d_ws, size_t ws_size,
                              hipStream_t stream) {
    const float* x   = (const float*)d_in[0];   // [1,2048,1024]
    const float* Wr  = (const float*)d_in[1];   // [8,1024]
    const float* Wfc = (const float*)d_in[2];   // [8,2048,1024]
    const float* Wp  = (const float*)d_in[3];   // [8,1024,2048]
    float* out = (float*)d_out;                 // 2048*1024 output + 1 aux_loss

    // Workspace layout:
    //   [0)      dense combine weights [2048][8] f32   (64 KB, fallback path)
    //   [64K)    cnt[8] (256B slot) | idx[8][2048] i32 | w[8][2048] f32
    //   [..)     x bf16 [2048][1024] | Wfc bf16 | Wp bf16
    char* ws = (char*)d_ws;
    const size_t off_wg  = 0;
    const size_t off_cnt = 65536;
    const size_t off_idx = off_cnt + 256;
    const size_t off_wl  = off_idx + (size_t)NEXP * TOK * 4;
    const size_t off_x   = off_wl  + (size_t)NEXP * TOK * 4;
    const size_t off_wfc = off_x   + (size_t)TOK * DIM * 2;
    const size_t off_wp  = off_wfc + (size_t)NEXP * HID * DIM * 2;
    const size_t ws_need = off_wp  + (size_t)NEXP * DIM * HID * 2;

    float*  wg    = (float*)(ws + off_wg);
    int*    cnt   = (int*)(ws + off_cnt);
    int*    idxl  = (int*)(ws + off_idx);
    float*  wl    = (float*)(ws + off_wl);
    __bf16* x16   = (__bf16*)(ws + off_x);
    __bf16* wfc16 = (__bf16*)(ws + off_wfc);
    __bf16* wp16  = (__bf16*)(ws + off_wp);

    const bool pre = (ws_size >= ws_need);   // deterministic across calls

    moe_zero_kernel<<<(out_size + 255) / 256, 256, 0, stream>>>(out, out_size);
    if (pre)  // zero the per-expert counters before the router fills lists
        moe_zero_kernel<<<1, 256, 0, stream>>>((float*)cnt, NEXP);
    moe_router_kernel<<<TOK / 8, 256, 0, stream>>>(x, Wr, wg, cnt, idxl, wl,
                                                   pre ? 1 : 0);
    if (pre) {
        const int nx = TOK * DIM / 4;
        const int nw = NEXP * HID * DIM / 4;
        moe_cvt_kernel<<<(nx + 255) / 256, 256, 0, stream>>>(x,   x16,   nx);
        moe_cvt_kernel<<<(nw + 255) / 256, 256, 0, stream>>>(Wfc, wfc16, nw);
        moe_cvt_kernel<<<(nw + 255) / 256, 256, 0, stream>>>(Wp,  wp16,  nw);
        moe_expert_kernel<true><<<dim3(TOK / TM, NEXP), 256, 0, stream>>>(
                x, x16, Wfc, Wp, wfc16, wp16, wg, cnt, idxl, wl, out);
    } else {
        moe_expert_kernel<false><<<dim3(TOK / TM, NEXP), 256, 0, stream>>>(
                x, x16, Wfc, Wp, wfc16, wp16, wg, cnt, idxl, wl, out);
    }
}
